// TransformerBlock_5342939316975
// MI455X (gfx1250) — compile-verified, matrix operand order checked by reading, not run on hardware
//
#include <hip/hip_runtime.h>
#include <hip/hip_bf16.h>

#define BB 2
#define TT 2048
#define DD 1024
#define HH 16
#define HD 64
#define NN (BB*TT)
#define DFF 4096
#define TILES 8
#define EPSF 1e-5f

typedef __bf16 bf16_t;
typedef __attribute__((ext_vector_type(16))) __bf16 bf16x16;
typedef __attribute__((ext_vector_type(8)))  __bf16 bf16x8;
typedef __attribute__((ext_vector_type(8)))  float  v8f;

__device__ __forceinline__ v8f zero8() {
  v8f z = {0.f,0.f,0.f,0.f,0.f,0.f,0.f,0.f};
  return z;
}

// Load a 16x32 bf16 A/B fragment slice for this lane from a row-major matrix.
// Lane L holds row (L&15); element e -> k = k0 + (e&7) + 8*hi + (e>=8 ? 16 : 0).
__device__ __forceinline__ bf16x16 load_frag(const bf16_t* row, int k0, int hi) {
  bf16x8 lo = *reinterpret_cast<const bf16x8*>(row + k0 + hi*8);
  bf16x8 hv = *reinterpret_cast<const bf16x8*>(row + k0 + 16 + hi*8);
  return __builtin_shufflevector(lo, hv, 0,1,2,3,4,5,6,7,8,9,10,11,12,13,14,15);
}

__device__ __forceinline__ v8f wmma_bf16(bf16x16 a, bf16x16 b, v8f c) {
  return __builtin_amdgcn_wmma_f32_16x16x32_bf16(false, a, false, b, (short)0, c, false, false);
}

__device__ __forceinline__ float rowmax16(float v) {
#pragma unroll
  for (int m = 1; m < 16; m <<= 1) v = fmaxf(v, __shfl_xor(v, m, 32));
  return v;
}
__device__ __forceinline__ float rowsum16(float v) {
#pragma unroll
  for (int m = 1; m < 16; m <<= 1) v += __shfl_xor(v, m, 32);
  return v;
}

// ---- Async global->LDS staging of one 64(cols) x 32(k) bf16 B-tile ----
// Tile layout in LDS: tile[m][k], row stride 32 halfs (64B). Each wave stages
// its 16 rows (wid*16..+15) with 2 async b128 ops (32 lanes x 16B each).
__device__ __forceinline__ void stage_tileB(const bf16_t* __restrict__ W, size_t ldb,
                                            int col0, int k0, bf16_t* sbuf,
                                            int wid, int lane) {
  int mloc = lane >> 2, chunk = lane & 3;
#pragma unroll
  for (int j = 0; j < 2; ++j) {
    int m = wid*16 + mloc + j*8;
    unsigned dst = (unsigned)(uintptr_t)(sbuf + m*32 + chunk*8);
    const bf16_t* src = W + (size_t)(col0 + m)*ldb + k0 + chunk*8;
    asm volatile("global_load_async_to_lds_b128 %0, %1, off"
                 :: "v"(dst), "v"(src) : "memory");
  }
}

// ---- 4-wave cooperative GEMM mainloop: block computes 64 rows x 64 cols.
// B tile double-buffered through LDS via async loads; A fragments direct global.
// acc[cc] accumulates this wave's 16x16 tile at cols col0+cc*16.
__device__ __forceinline__ void gemm_mainloop(const bf16_t* __restrict__ arow,
                                              const bf16_t* __restrict__ W, size_t ldb,
                                              int col0, int K, bf16_t* sB, v8f acc[4]) {
  int tid = threadIdx.x, wid = tid >> 5, lane = tid & 31;
  int l16 = lane & 15, hi = lane >> 4;
  stage_tileB(W, ldb, col0, 0, sB, wid, lane);
  asm volatile("s_wait_asynccnt 0" ::: "memory");
  __syncthreads();
  int nk = K >> 5;
  for (int kk = 0; kk < nk; ++kk) {
    int buf = kk & 1;
    bf16_t* cur = sB + buf*(64*32);
    if (kk + 1 < nk) stage_tileB(W, ldb, col0, (kk+1)*32, sB + (buf^1)*(64*32), wid, lane);
    bf16x16 af = load_frag(arow, kk*32, hi);
#pragma unroll
    for (int cc = 0; cc < 4; ++cc) {
      bf16x16 bfr = load_frag(cur + (cc*16 + l16)*32, 0, hi);
      acc[cc] = wmma_bf16(af, bfr, acc[cc]);
    }
    asm volatile("s_wait_asynccnt 0" ::: "memory");
    __syncthreads();
  }
}

// ---------------- LayerNorm: f32 [rows, DD] -> bf16 ----------------
__global__ void ln_kernel(const float* __restrict__ x, const float* __restrict__ w,
                          const float* __restrict__ bv, bf16_t* __restrict__ out) {
  int row = blockIdx.x;
  int t = threadIdx.x;
  const float* xr = x + (size_t)row * DD;
  float v[4]; float s = 0.f;
#pragma unroll
  for (int i = 0; i < 4; ++i) { v[i] = xr[t + i*256]; s += v[i]; }
  __shared__ float red[256];
  red[t] = s; __syncthreads();
  for (int off = 128; off > 0; off >>= 1) { if (t < off) red[t] += red[t+off]; __syncthreads(); }
  float mean = red[0] * (1.f/DD);
  __syncthreads();
  float s2 = 0.f;
#pragma unroll
  for (int i = 0; i < 4; ++i) { float d = v[i]-mean; s2 += d*d; }
  red[t] = s2; __syncthreads();
  for (int off = 128; off > 0; off >>= 1) { if (t < off) red[t] += red[t+off]; __syncthreads(); }
  float rstd = rsqrtf(red[0]*(1.f/DD) + EPSF);
  bf16_t* orow = out + (size_t)row*DD;
#pragma unroll
  for (int i = 0; i < 4; ++i) { int c = t + i*256; orow[c] = (bf16_t)((v[i]-mean)*rstd*w[c] + bv[c]); }
}

// ---------------- f32 -> bf16 conversion ----------------
__global__ void cvt_kernel(const float* __restrict__ src, bf16_t* __restrict__ dst, size_t n) {
  size_t i = (size_t)blockIdx.x*blockDim.x + threadIdx.x;
  size_t st = (size_t)gridDim.x*blockDim.x;
  for (; i < n; i += st) dst[i] = (bf16_t)src[i];
}

// ---------------- down_w summed over tiles -> bf16 [DD, DFF] ----------------
__global__ void dwsum_kernel(const float* __restrict__ dw, bf16_t* __restrict__ wdn) {
  size_t n = (size_t)DD*DFF;
  size_t i = (size_t)blockIdx.x*blockDim.x + threadIdx.x;
  size_t st = (size_t)gridDim.x*blockDim.x;
  for (; i < n; i += st) {
    float s = 0.f;
#pragma unroll
    for (int t = 0; t < TILES; ++t) s += dw[(size_t)t*DD*DFF + i];
    wdn[i] = (bf16_t)s;
  }
}
__global__ void dbsum_kernel(const float* __restrict__ db, float* __restrict__ dbs) {
  int c = blockIdx.x*blockDim.x + threadIdx.x;
  if (c < DD) {
    float s = 0.f;
#pragma unroll
    for (int t = 0; t < TILES; ++t) s += db[t*DD + c];
    dbs[c] = s;
  }
}

// ---------------- QKV GEMM: nrm1 @ in_proj_w^T + b, scatter to q/k/vt ----------------
__global__ void __launch_bounds__(128)
qkv_gemm(const bf16_t* __restrict__ A, const bf16_t* __restrict__ W,
         const float* __restrict__ bias,
         bf16_t* __restrict__ q, bf16_t* __restrict__ kk, bf16_t* __restrict__ vt) {
  __shared__ __attribute__((aligned(16))) bf16_t sB[2*64*32];
  int tid = threadIdx.x, wid = tid >> 5, lane = tid & 31, l16 = lane & 15, hi = lane >> 4;
  int col0 = blockIdx.x * 64, row0 = blockIdx.y * 64 + wid*16;
  v8f acc[4] = {zero8(), zero8(), zero8(), zero8()};
  const bf16_t* arow = A + (size_t)(row0 + l16) * DD;
  gemm_mainloop(arow, W, DD, col0, DD, sB, acc);
#pragma unroll
  for (int cc = 0; cc < 4; ++cc) {
#pragma unroll
    for (int r = 0; r < 8; ++r) {
      int m = col0 + cc*16 + l16;
      int n = row0 + r + 8*hi;
      float val = acc[cc][r] + bias[m];
      int which = m >> 10, hcol = m & 1023, head = hcol >> 6, hd = hcol & 63;
      int b = n >> 11, tpos = n & 2047;
      if (which == 2) {
        // V stored transposed per head: [B,H,HD,T]
        vt[((size_t)(b*HH + head)*HD + hd)*TT + tpos] = (bf16_t)val;
      } else {
        size_t idx = ((size_t)(b*HH + head)*TT + tpos)*HD + hd;
        if (which == 0) q[idx] = (bf16_t)val; else kk[idx] = (bf16_t)val;
      }
    }
  }
}

// ---------------- Flash-style causal attention: 1 wave per 16-query tile ----------------
__global__ void attn_kernel(const bf16_t* __restrict__ Q, const bf16_t* __restrict__ Km,
                            const bf16_t* __restrict__ Vt, bf16_t* __restrict__ ctx) {
  int qt = blockIdx.x, h = blockIdx.y, b = blockIdx.z;
  int lane = threadIdx.x, l16 = lane & 15, hi = lane >> 4;
  const bf16_t* Qb = Q  + (size_t)(b*HH + h) * TT * HD;
  const bf16_t* Kb = Km + (size_t)(b*HH + h) * TT * HD;
  const bf16_t* Vb = Vt + (size_t)(b*HH + h) * HD * TT;
  const bf16_t* qrow = Qb + (size_t)(qt*16 + l16) * HD;
  bf16x16 qf0 = load_frag(qrow, 0, hi);
  bf16x16 qf1 = load_frag(qrow, 32, hi);
  float mrow[8], lrow[8];
  v8f o[4] = {zero8(), zero8(), zero8(), zero8()};
#pragma unroll
  for (int r = 0; r < 8; ++r) { mrow[r] = -3.0e38f; lrow[r] = 0.f; }
  __shared__ __attribute__((aligned(16))) bf16_t sP[16*40];  // 16 q rows x 32 k, padded stride 40
  int kmax = qt*16 + 16;
  int nkb = (kmax + 31) >> 5;
  for (int kb = 0; kb < nkb; ++kb) {
    int kbase = kb * 32;
    v8f s[2];
#pragma unroll
    for (int sub = 0; sub < 2; ++sub) {
      int kcb = kbase + sub*16;
      if (kcb < kmax) {
        const bf16_t* krow = Kb + (size_t)(kcb + l16) * HD;
        bf16x16 b0 = load_frag(krow, 0, hi);
        bf16x16 b1 = load_frag(krow, 32, hi);
        v8f a = zero8();
        a = wmma_bf16(qf0, b0, a);
        a = wmma_bf16(qf1, b1, a);
        int kg = kcb + l16;
#pragma unroll
        for (int r = 0; r < 8; ++r) {
          int qg = qt*16 + r + 8*hi;
          s[sub][r] = (kg <= qg) ? a[r] * 0.125f : -3.0e38f;   // 1/sqrt(64)
        }
      } else {
#pragma unroll
        for (int r = 0; r < 8; ++r) s[sub][r] = -3.0e38f;
      }
    }
    float scl[8];
#pragma unroll
    for (int r = 0; r < 8; ++r) {
      float tm = fmaxf(rowmax16(s[0][r]), rowmax16(s[1][r]));
      float mnew = fmaxf(mrow[r], tm);
      scl[r] = __expf(mrow[r] - mnew);
      mrow[r] = mnew;
      lrow[r] *= scl[r];
    }
#pragma unroll
    for (int cc = 0; cc < 4; ++cc)
#pragma unroll
      for (int r = 0; r < 8; ++r) o[cc][r] *= scl[r];
    __syncthreads();
#pragma unroll
    for (int sub = 0; sub < 2; ++sub) {
#pragma unroll
      for (int r = 0; r < 8; ++r) {
        float p = __expf(s[sub][r] - mrow[r]);
        lrow[r] += rowsum16(p);
        sP[(r + 8*hi)*40 + sub*16 + l16] = (bf16_t)p;  // D-layout -> [q][k] in LDS
      }
    }
    __syncthreads();
    bf16x16 pf = load_frag(sP + l16*40, 0, hi);        // reload as A-fragment
#pragma unroll
    for (int cc = 0; cc < 4; ++cc) {
      const bf16_t* vrow = Vb + (size_t)(cc*16 + l16) * TT + kbase;
      bf16x16 vf = load_frag(vrow, 0, hi);
      o[cc] = wmma_bf16(pf, vf, o[cc]);
    }
  }
#pragma unroll
  for (int r = 0; r < 8; ++r) lrow[r] = 1.f / lrow[r];
#pragma unroll
  for (int cc = 0; cc < 4; ++cc)
#pragma unroll
    for (int r = 0; r < 8; ++r) {
      int qg = qt*16 + r + 8*hi;
      ctx[((size_t)(b*TT + qg))*DD + h*HD + cc*16 + l16] = (bf16_t)(o[cc][r] * lrow[r]);
    }
}

// ---------------- out-proj GEMM + residual -> x1 (f32) ----------------
__global__ void __launch_bounds__(128)
outproj_gemm(const bf16_t* __restrict__ A, const bf16_t* __restrict__ W,
             const float* __restrict__ bias, const float* __restrict__ xres,
             float* __restrict__ x1) {
  __shared__ __attribute__((aligned(16))) bf16_t sB[2*64*32];
  int tid = threadIdx.x, wid = tid >> 5, lane = tid & 31, l16 = lane & 15, hi = lane >> 4;
  int col0 = blockIdx.x * 64, row0 = blockIdx.y * 64 + wid*16;
  v8f acc[4] = {zero8(),zero8(),zero8(),zero8()};
  const bf16_t* arow = A + (size_t)(row0 + l16)*DD;
  gemm_mainloop(arow, W, DD, col0, DD, sB, acc);
#pragma unroll
  for (int cc = 0; cc < 4; ++cc)
#pragma unroll
    for (int r = 0; r < 8; ++r) {
      int m = col0 + cc*16 + l16, n = row0 + r + 8*hi;
      size_t idx = (size_t)n*DD + m;
      x1[idx] = acc[cc][r] + bias[m] + xres[idx];
    }
}

// ---------------- routing: logits, argmax, one-hot gate, counts ----------------
__global__ void route_kernel(const bf16_t* __restrict__ h, const float* __restrict__ gw,
                             const float* __restrict__ gb, int* __restrict__ route,
                             float* __restrict__ gate_out, int* __restrict__ counts) {
  int n = blockIdx.x*blockDim.x + threadIdx.x;
  if (n >= NN) return;
  const bf16_t* hr = h + (size_t)n*DD;
  float best = -3.0e38f; int bi = 0;
  for (int t = 0; t < TILES; ++t) {
    float s = gb[t];
    const float* gr = gw + (size_t)t*DD;
    for (int c = 0; c < DD; ++c) s += (float)hr[c] * gr[c];
    if (s > best) { best = s; bi = t; }
  }
  route[n] = bi;
#pragma unroll
  for (int t = 0; t < TILES; ++t) gate_out[(size_t)n*TILES + t] = (t == bi) ? 1.f : 0.f;
  atomicAdd(&counts[bi], 1);
}

__global__ void init_kernel(int* counts) { if (threadIdx.x < TILES) counts[threadIdx.x] = 0; }

__global__ void scan_kernel(const int* __restrict__ counts, int* __restrict__ offs,
                            int* __restrict__ cursor) {
  if (threadIdx.x == 0) {
    int acc = 0;
    for (int t = 0; t < TILES; ++t) { offs[t] = acc; cursor[t] = acc; acc += counts[t]; }
  }
}

__global__ void gather_kernel(const int* __restrict__ route, int* __restrict__ cursor,
                              int* __restrict__ perm) {
  int n = blockIdx.x*blockDim.x + threadIdx.x;
  if (n < NN) { int slot = atomicAdd(&cursor[route[n]], 1); perm[slot] = n; }
}

// ---------------- routed up GEMM (+bias, ReLU) over per-tile buckets ----------------
__global__ void __launch_bounds__(128)
up_gemm(const bf16_t* __restrict__ A, const bf16_t* __restrict__ Wup,
        const float* __restrict__ upb, const int* __restrict__ counts,
        const int* __restrict__ offs, const int* __restrict__ perm,
        bf16_t* __restrict__ hidden) {
  int t = blockIdx.z;
  int cnt = counts[t];
  int rt = blockIdx.y;
  if (rt*64 >= cnt) return;          // uniform across block
  __shared__ __attribute__((aligned(16))) bf16_t sB[2*64*32];
  int off = offs[t];
  int tid = threadIdx.x, wid = tid >> 5, lane = tid & 31, l16 = lane & 15, hi = lane >> 4;
  int col0 = blockIdx.x*64;
  int ra = rt*64 + wid*16 + l16;
  int tokA = perm[off + ((ra < cnt) ? ra : 0)];
  const bf16_t* arow = A + (size_t)tokA*DD;
  const bf16_t* Wt = Wup + (size_t)t*DFF*DD;
  v8f acc[4] = {zero8(),zero8(),zero8(),zero8()};
  gemm_mainloop(arow, Wt, DD, col0, DD, sB, acc);
  int tokr[8];
#pragma unroll
  for (int r = 0; r < 8; ++r) {
    int ridx = rt*64 + wid*16 + r + 8*hi;
    tokr[r] = (ridx < cnt) ? perm[off + ridx] : -1;
  }
#pragma unroll
  for (int cc = 0; cc < 4; ++cc)
#pragma unroll
    for (int r = 0; r < 8; ++r) {
      if (tokr[r] >= 0) {
        int m = col0 + cc*16 + l16;
        float v = acc[cc][r] + upb[(size_t)t*DFF + m];
        hidden[(size_t)tokr[r]*DFF + m] = (bf16_t)fmaxf(v, 0.f);
      }
    }
}

// ---------------- down GEMM (+summed bias, residual) -> d_out f32 ----------------
__global__ void __launch_bounds__(128)
down_gemm(const bf16_t* __restrict__ A, const bf16_t* __restrict__ W,
          const float* __restrict__ dbs, const float* __restrict__ x1,
          float* __restrict__ out) {
  __shared__ __attribute__((aligned(16))) bf16_t sB[2*64*32];
  int tid = threadIdx.x, wid = tid >> 5, lane = tid & 31, l16 = lane & 15, hi = lane >> 4;
  int col0 = blockIdx.x*64, row0 = blockIdx.y*64 + wid*16;
  v8f acc[4] = {zero8(),zero8(),zero8(),zero8()};
  const bf16_t* arow = A + (size_t)(row0 + l16)*DFF;
  gemm_mainloop(arow, W, DFF, col0, DFF, sB, acc);
#pragma unroll
  for (int cc = 0; cc < 4; ++cc)
#pragma unroll
    for (int r = 0; r < 8; ++r) {
      int m = col0 + cc*16 + l16, n = row0 + r + 8*hi;
      size_t idx = (size_t)n*DD + m;
      out[idx] = acc[cc][r] + dbs[m] + x1[idx];
    }
}

extern "C" void kernel_launch(void* const* d_in, const int* in_sizes, int n_in,
                              void* d_out, int out_size, void* d_ws, size_t ws_size,
                              hipStream_t stream) {
  const float* x    = (const float*)d_in[0];
  const float* ln1w = (const float*)d_in[1];
  const float* ln1b = (const float*)d_in[2];
  const float* ln2w = (const float*)d_in[3];
  const float* ln2b = (const float*)d_in[4];
  const float* inw  = (const float*)d_in[5];
  const float* inb  = (const float*)d_in[6];
  const float* outw = (const float*)d_in[7];
  const float* outb = (const float*)d_in[8];
  const float* gw   = (const float*)d_in[9];
  const float* gb   = (const float*)d_in[10];
  const float* upw  = (const float*)d_in[11];
  const float* upb  = (const float*)d_in[12];
  const float* dww  = (const float*)d_in[13];
  const float* dwb  = (const float*)d_in[14];

  char* wsp = (char*)d_ws;
  auto alloc = [&](size_t bytes) -> void* {
    void* p = (void*)wsp;
    wsp += (bytes + 255) & ~(size_t)255;
    return p;
  };
  bf16_t* nrm1 = (bf16_t*)alloc((size_t)NN*DD*2);
  bf16_t* wqkv = (bf16_t*)alloc((size_t)3*DD*DD*2);
  bf16_t* wout = (bf16_t*)alloc((size_t)DD*DD*2);
  bf16_t* qb   = (bf16_t*)alloc((size_t)NN*DD*2);
  bf16_t* kb   = (bf16_t*)alloc((size_t)NN*DD*2);
  bf16_t* vtb  = (bf16_t*)alloc((size_t)NN*DD*2);
  bf16_t* ctx  = (bf16_t*)alloc((size_t)NN*DD*2);
  float*  x1   = (float*)alloc((size_t)NN*DD*4);
  bf16_t* nrm2 = (bf16_t*)alloc((size_t)NN*DD*2);
  bf16_t* wup  = (bf16_t*)alloc((size_t)TILES*DFF*DD*2);
  bf16_t* wdn  = (bf16_t*)alloc((size_t)DD*DFF*2);
  float*  dbs  = (float*)alloc((size_t)DD*4);
  bf16_t* hid  = (bf16_t*)alloc((size_t)NN*DFF*2);
  int* route   = (int*)alloc((size_t)NN*4);
  int* perm    = (int*)alloc((size_t)NN*4);
  int* counts  = (int*)alloc(TILES*4);
  int* offs    = (int*)alloc(TILES*4);
  int* cursor  = (int*)alloc(TILES*4);

  float* out_x    = (float*)d_out;
  float* out_gate = out_x + (size_t)NN*DD;

  init_kernel<<<1, 32, 0, stream>>>(counts);
  cvt_kernel<<<2048, 256, 0, stream>>>(inw,  wqkv, (size_t)3*DD*DD);
  cvt_kernel<<<1024, 256, 0, stream>>>(outw, wout, (size_t)DD*DD);
  cvt_kernel<<<4096, 256, 0, stream>>>(upw,  wup,  (size_t)TILES*DFF*DD);
  dwsum_kernel<<<4096, 256, 0, stream>>>(dww, wdn);
  dbsum_kernel<<<4, 256, 0, stream>>>(dwb, dbs);
  ln_kernel<<<NN, 256, 0, stream>>>(x, ln1w, ln1b, nrm1);
  qkv_gemm<<<dim3(3*DD/64, NN/64), 128, 0, stream>>>(nrm1, wqkv, inb, qb, kb, vtb);
  attn_kernel<<<dim3(TT/16, HH, BB), 32, 0, stream>>>(qb, kb, vtb, ctx);
  outproj_gemm<<<dim3(DD/64, NN/64), 128, 0, stream>>>(ctx, wout, outb, x, x1);
  ln_kernel<<<NN, 256, 0, stream>>>(x1, ln2w, ln2b, nrm2);
  route_kernel<<<NN/256, 256, 0, stream>>>(nrm2, gw, gb, route, out_gate, counts);
  scan_kernel<<<1, 32, 0, stream>>>(counts, offs, cursor);
  gather_kernel<<<NN/256, 256, 0, stream>>>(route, cursor, perm);
  up_gemm<<<dim3(DFF/64, NN/64, TILES), 128, 0, stream>>>(nrm2, wup, upb, counts, offs, perm, hid);
  down_gemm<<<dim3(DD/64, NN/64), 128, 0, stream>>>(hid, wdn, dbs, x1, out_x);
  (void)in_sizes; (void)n_in; (void)out_size; (void)ws_size;
}
